// FakeFlexOlmoSparseMlp_11793980194917
// MI455X (gfx1250) — matrix operand order repaired and checked by reading, not compiled, not run on hardware
//
#include <hip/hip_runtime.h>
#include <hip/hip_bf16.h>

// Problem constants (match reference)
#define HDIM 1024
#define NEXP 8
#define NTOK 8192          // B*S = 4*2048
#define MTILE 32           // tokens per workgroup in FFN kernel
#define PITCH 1032         // LDS row pitch in bf16 elems (512+2+2 DWORD layout)

typedef __attribute__((ext_vector_type(16))) __bf16 v16bf;
typedef __attribute__((ext_vector_type(8)))  __bf16 v8bf;
typedef __attribute__((ext_vector_type(8)))  float  v8f;
typedef __attribute__((ext_vector_type(4)))  unsigned int u32x4;
typedef __attribute__((ext_vector_type(8)))  int i32x8;
typedef __attribute__((ext_vector_type(4)))  int i32x4;

#if __has_builtin(__builtin_amdgcn_tensor_load_to_lds) && \
    __has_builtin(__builtin_amdgcn_s_wait_tensorcnt)
#define USE_TDM 1
#endif

// LDS column mapping: TDM inserts a 2-DWORD pad after every 256 DWORDs
// (1024 B), i.e. a 4-half gap at element column 512 and at end of row.
__device__ __forceinline__ int scol(int c) { return c + ((c >= 512) ? 4 : 0); }

// A fragment (16x32 bf16, ISA layout): lane holds K {kb..kb+7, kb+16..kb+23}
__device__ __forceinline__ v16bf load_frag_a(const __bf16* base, int k0) {
    const __bf16* p = base + scol(k0);
    v8bf lo = *(const v8bf*)p;
    v8bf hi = *(const v8bf*)(p + 16);
    v16bf a;
#pragma unroll
    for (int i = 0; i < 8; ++i) { a[i] = lo[i]; a[i + 8] = hi[i]; }
    return a;
}

// B fragment (32x16 bf16, ISA layout): lane holds K {kb..kb+7, kb+16..kb+23}
__device__ __forceinline__ v16bf load_frag_b(const __bf16* brow, int k0) {
    v8bf lo = *(const v8bf*)(brow + k0);
    v8bf hi = *(const v8bf*)(brow + k0 + 16);
    v16bf b;
#pragma unroll
    for (int i = 0; i < 8; ++i) { b[i] = lo[i]; b[i + 8] = hi[i]; }
    return b;
}

__device__ __forceinline__ v8f wmma_bf16(v16bf a, v16bf b, v8f c) {
    return __builtin_amdgcn_wmma_f32_16x16x32_bf16(false, a, false, b,
                                                   (short)0, c, false, false);
}

// ---------------------------------------------------------------------------
// Kernel 1: convert expert weights f32 -> bf16 into workspace.
// ---------------------------------------------------------------------------
__global__ __launch_bounds__(256) void cvt_kernel(
    const float* __restrict__ W1, const float* __restrict__ W2,
    __bf16* __restrict__ W1b, __bf16* __restrict__ W2b, int n)
{
    int i = blockIdx.x * blockDim.x + threadIdx.x;
    int stride = gridDim.x * blockDim.x;
    for (; i < n; i += stride) {
        W1b[i] = (__bf16)W1[i];
        W2b[i] = (__bf16)W2[i];
    }
}

// ---------------------------------------------------------------------------
// Kernel 2: router. One wave32 per token: logits, softmax, top-2 renorm.
// Also emits bf16 copy of hidden (X) and writes the residual into out.
// ---------------------------------------------------------------------------
__global__ __launch_bounds__(256) void router_kernel(
    const float* __restrict__ hidden, const float* __restrict__ router_w,
    float* __restrict__ out, float* __restrict__ probs_out,
    __bf16* __restrict__ Xb, float* __restrict__ wcomb)
{
    const int tok  = (blockIdx.x * blockDim.x + threadIdx.x) >> 5;
    const int lane = threadIdx.x & 31;
    if (tok >= NTOK) return;

    const float* x = hidden + (size_t)tok * HDIM;
    float acc[NEXP];
#pragma unroll
    for (int e = 0; e < NEXP; ++e) acc[e] = 0.0f;

    for (int h = lane; h < HDIM; h += 32) {
        float xv = x[h];
        out[(size_t)tok * HDIM + h] = xv;           // residual base
        Xb[(size_t)tok * HDIM + h]  = (__bf16)xv;   // bf16 activation copy
#pragma unroll
        for (int e = 0; e < NEXP; ++e) acc[e] += xv * router_w[e * HDIM + h];
    }
#pragma unroll
    for (int off = 16; off >= 1; off >>= 1) {
#pragma unroll
        for (int e = 0; e < NEXP; ++e) acc[e] += __shfl_down(acc[e], off, 32);
    }

    if (lane == 0) {
        float mx = acc[0];
#pragma unroll
        for (int e = 1; e < NEXP; ++e) mx = fmaxf(mx, acc[e]);
        float p[NEXP], s = 0.0f;
#pragma unroll
        for (int e = 0; e < NEXP; ++e) { p[e] = __expf(acc[e] - mx); s += p[e]; }
        float inv = 1.0f / s;
        int i1 = 0; float v1 = -1.0f;
#pragma unroll
        for (int e = 0; e < NEXP; ++e) {
            p[e] *= inv;
            probs_out[(size_t)tok * NEXP + e] = p[e];
            if (p[e] > v1) { v1 = p[e]; i1 = e; }
        }
        int i2 = 0; float v2 = -1.0f;
#pragma unroll
        for (int e = 0; e < NEXP; ++e)
            if (e != i1 && p[e] > v2) { v2 = p[e]; i2 = e; }
        float rn = 1.0f / (v1 + v2);
#pragma unroll
        for (int e = 0; e < NEXP; ++e) wcomb[(size_t)tok * NEXP + e] = 0.0f;
        wcomb[(size_t)tok * NEXP + i1] = v1 * rn;
        wcomb[(size_t)tok * NEXP + i2] = v2 * rn;
    }
}

// ---------------------------------------------------------------------------
// Kernel 3: fused dense-MoE FFN with 2x4 register blocking per wave
// (each wave owns a 32x64 output block -> 8 WMMAs per 12 b128 loads).
// One workgroup = (32-token tile, expert e); X tile staged via TDM.
// ---------------------------------------------------------------------------
__global__ __launch_bounds__(256) void moe_ffn_kernel(
    const __bf16* __restrict__ Xb,  const __bf16* __restrict__ W1b,
    const float*  __restrict__ b1,  const __bf16* __restrict__ W2b,
    const float*  __restrict__ b2,  const float*  __restrict__ wcomb,
    float* __restrict__ out)
{
    __shared__ __bf16 sX [MTILE * PITCH];
    __shared__ __bf16 sH1[MTILE * PITCH];

    const int e    = blockIdx.y;
    const int t    = blockIdx.x;            // token tile index
    const int tid  = threadIdx.x;
    const int lane = tid & 31;
    const int wave = tid >> 5;
    const int lm   = lane & 15;
    const int hiH  = (lane < 16) ? 0 : 1;   // lane half selector

    const __bf16* xsrc = Xb + (size_t)t * MTILE * HDIM;

#ifdef USE_TDM
    // ---- Tensor Data Mover: DMA the 32x1024 bf16 tile into LDS with
    //      hardware row padding (2 DWORDs per 256 DWORDs -> pitch 1032). ----
    if (wave == 0) {
        unsigned long long ga = (unsigned long long)(uintptr_t)xsrc;
        u32x4 g0;
        g0[0] = 1u;                                   // count=1, user descriptor
        g0[1] = (unsigned)(uintptr_t)sX;              // lds_addr (low 32 = LDS offset)
        g0[2] = (unsigned)ga;                         // global_addr[31:0]
        g0[3] = ((unsigned)(ga >> 32) & 0x01FFFFFFu)  // global_addr[56:32]
                | (2u << 30);                         // type = 2 (image)
        i32x8 g1;
        g1[0] = (1 << 16)    // data_size = 2 bytes
              | (1 << 20)    // pad_enable
              | (7 << 22)    // pad_interval: 256 DWORDs
              | (1 << 25);   // pad_amount: 2 DWORDs
        g1[1] = (int)(1024u << 16);   // tensor_dim0[15:0]=1024 (atomic_addr=0)
        g1[2] = (int)(32u   << 16);   // tensor_dim0 hi=0 | tensor_dim1=32 rows
        g1[3] = (int)(1024u << 16);   // tensor_dim1 hi=0 | tile_dim0=1024
        g1[4] = 32;                   // tile_dim1=32, tile_dim2=0
        g1[5] = 1024;                 // tensor_dim0_stride low32
        g1[6] = 0;                    // stride hi, tensor_dim1_stride lo
        g1[7] = 0;
        i32x4 gz = {0, 0, 0, 0};
#if __has_include(<hip/amd_detail/amd_gfx1250_TDM.h>)
        i32x8 gz8 = {0, 0, 0, 0, 0, 0, 0, 0};
        __builtin_amdgcn_tensor_load_to_lds(g0, g1, gz, gz, gz8, 0);
#else
        __builtin_amdgcn_tensor_load_to_lds(g0, g1, gz, gz, 0);
#endif
        __builtin_amdgcn_s_wait_tensorcnt(0);
    }
#else
    for (int c = tid; c < MTILE * (HDIM / 8); c += 256) {
        int row = c >> 7;
        int col = (c & 127) << 3;
        *(v8bf*)&sX[row * PITCH + scol(col)] = *(const v8bf*)&xsrc[row * HDIM + col];
    }
#endif
    __syncthreads();

    // ================= stage 1: H1 = relu(X @ W1^T + b1) =================
    {
        const __bf16* We = W1b + (size_t)e * HDIM * HDIM;
        for (int g = wave; g < HDIM / 64; g += 8) {
            int nc[4];
            const __bf16* bp[4];
            v8f acc[2][4];
#pragma unroll
            for (int j = 0; j < 4; ++j) {
                nc[j] = g * 64 + j * 16 + lm;
                bp[j] = We + (size_t)nc[j] * HDIM + hiH * 8;
                float bias = b1[e * HDIM + nc[j]];
#pragma unroll
                for (int r = 0; r < 8; ++r) { acc[0][j][r] = bias; acc[1][j][r] = bias; }
            }
            const __bf16* a0p = &sX[lm * PITCH + hiH * 8];
            const __bf16* a1p = &sX[(16 + lm) * PITCH + hiH * 8];
#pragma unroll 2
            for (int k0 = 0; k0 < HDIM; k0 += 32) {
                if ((k0 & 127) == 0) {
#pragma unroll
                    for (int j = 0; j < 4; ++j)
                        __builtin_prefetch(bp[j] + k0 + 256, 0, 3);
                }
                v16bf af[2];
                af[0] = load_frag_a(a0p, k0);
                af[1] = load_frag_a(a1p, k0);
                v16bf bf[4];
#pragma unroll
                for (int j = 0; j < 4; ++j) bf[j] = load_frag_b(bp[j], k0);
#pragma unroll
                for (int mi = 0; mi < 2; ++mi)
#pragma unroll
                    for (int j = 0; j < 4; ++j)
                        acc[mi][j] = wmma_bf16(af[mi], bf[j], acc[mi][j]);
            }
            // ReLU + pack to bf16 in LDS (C-layout scatter)
#pragma unroll
            for (int mi = 0; mi < 2; ++mi) {
#pragma unroll
                for (int j = 0; j < 4; ++j) {
                    int sc = scol(nc[j]);
#pragma unroll
                    for (int r = 0; r < 8; ++r) {
                        int m = mi * 16 + r + hiH * 8;
                        sH1[m * PITCH + sc] = (__bf16)fmaxf(acc[mi][j][r], 0.0f);
                    }
                }
            }
        }
    }
    __syncthreads();

    // ============ stage 2: out += w * (H1 @ W2^T + b2) ============
    {
        const __bf16* We   = W2b + (size_t)e * HDIM * HDIM;
        const size_t  tok0 = (size_t)t * MTILE;
        for (int g = wave; g < HDIM / 64; g += 8) {
            int nc[4];
            const __bf16* bp[4];
            v8f acc[2][4];
#pragma unroll
            for (int j = 0; j < 4; ++j) {
                nc[j] = g * 64 + j * 16 + lm;
                bp[j] = We + (size_t)nc[j] * HDIM + hiH * 8;
                float bias = b2[e * HDIM + nc[j]];
#pragma unroll
                for (int r = 0; r < 8; ++r) { acc[0][j][r] = bias; acc[1][j][r] = bias; }
            }
            const __bf16* a0p = &sH1[lm * PITCH + hiH * 8];
            const __bf16* a1p = &sH1[(16 + lm) * PITCH + hiH * 8];
#pragma unroll 2
            for (int k0 = 0; k0 < HDIM; k0 += 32) {
                if ((k0 & 127) == 0) {
#pragma unroll
                    for (int j = 0; j < 4; ++j)
                        __builtin_prefetch(bp[j] + k0 + 256, 0, 3);
                }
                v16bf af[2];
                af[0] = load_frag_a(a0p, k0);
                af[1] = load_frag_a(a1p, k0);
                v16bf bf[4];
#pragma unroll
                for (int j = 0; j < 4; ++j) bf[j] = load_frag_b(bp[j], k0);
#pragma unroll
                for (int mi = 0; mi < 2; ++mi)
#pragma unroll
                    for (int j = 0; j < 4; ++j)
                        acc[mi][j] = wmma_bf16(af[mi], bf[j], acc[mi][j]);
            }
            // scale by per-token combine weight, accumulate into residual
#pragma unroll
            for (int mi = 0; mi < 2; ++mi) {
#pragma unroll
                for (int r = 0; r < 8; ++r) {
                    size_t tok = tok0 + mi * 16 + r + hiH * 8;
                    float  wgt = wcomb[tok * NEXP + e];
#pragma unroll
                    for (int j = 0; j < 4; ++j)
                        __hip_atomic_fetch_add(&out[tok * HDIM + nc[j]],
                                               acc[mi][j][r] * wgt,
                                               __ATOMIC_RELAXED,
                                               __HIP_MEMORY_SCOPE_AGENT);
                }
            }
        }
    }
}

// ---------------------------------------------------------------------------
// Launcher. ws layout: wcomb (256KB) | Xb (16MB) | W1b (16MB) | W2b (16MB)
// ---------------------------------------------------------------------------
extern "C" void kernel_launch(void* const* d_in, const int* in_sizes, int n_in,
                              void* d_out, int out_size, void* d_ws, size_t ws_size,
                              hipStream_t stream)
{
    const float* hidden   = (const float*)d_in[0];
    const float* router_w = (const float*)d_in[1];
    const float* W1       = (const float*)d_in[2];
    const float* b1       = (const float*)d_in[3];
    const float* W2       = (const float*)d_in[4];
    const float* b2       = (const float*)d_in[5];

    float* out   = (float*)d_out;                 // [NTOK*HDIM]
    float* probs = out + (size_t)NTOK * HDIM;     // [NTOK*NEXP]

    char*   ws    = (char*)d_ws;
    float*  wcomb = (float*)ws;   ws += sizeof(float)  * (size_t)NTOK * NEXP;
    __bf16* Xb    = (__bf16*)ws;  ws += sizeof(__bf16) * (size_t)NTOK * HDIM;
    __bf16* W1b   = (__bf16*)ws;  ws += sizeof(__bf16) * (size_t)NEXP * HDIM * HDIM;
    __bf16* W2b   = (__bf16*)ws;

    cvt_kernel<<<4096, 256, 0, stream>>>(W1, W2, W1b, W2b, NEXP * HDIM * HDIM);
    router_kernel<<<NTOK / 8, 256, 0, stream>>>(hidden, router_w, out, probs, Xb, wcomb);
    dim3 grid(NTOK / MTILE, NEXP);
    moe_ffn_kernel<<<grid, 256, 0, stream>>>(Xb, W1b, b1, W2b, b2, wcomb, out);
}
